// Attention_90168543412890
// MI455X (gfx1250) — compile-verified, matrix operand order checked by reading, not run on hardware
//
#include <hip/hip_runtime.h>
#include <hip/hip_bf16.h>
#include <stdint.h>

typedef __bf16 v16bf __attribute__((ext_vector_type(16)));
typedef float  v8f   __attribute__((ext_vector_type(8)));

constexpr int kB = 2, kS = 2048, kD = 1024, kH = 16, kHD = 64;
constexpr int kPS = kS + 4;   // attention LDS stride (floats); 2052 % 64 = 4 -> conflict-free
constexpr int kXP = kD + 4;   // A-strip LDS stride (floats); 1028 % 64 = 4; rows stay 16B-aligned

// ---- WMMA fragment index helpers (CDNA5 ISA 7.12.2, 16-bit A 16x32 layout) ----
__device__ __forceinline__ int frag_k(int e, int half) {
  // lanes 0-15: K = 0..7 (e=0..7), 16..23 (e=8..15); lanes 16-31: +8
  return (e & 7) + ((e >> 3) << 4) + (half << 3);
}

__device__ __forceinline__ v8f wmma_bf16(v16bf a, v16bf b, v8f c) {
  return __builtin_amdgcn_wmma_f32_16x16x32_bf16(
      /*neg_a=*/false, a, /*neg_b=*/false, b,
      /*c_mod=*/(short)0, c, /*reuse_a=*/false, /*reuse_b=*/false);
}

// A fragment from row-major bf16 global
__device__ __forceinline__ v16bf load_a_bf16(const __bf16* __restrict__ p, int row0, int ld,
                                             int koff, int lr, int half) {
  v16bf f;
#pragma unroll
  for (int e = 0; e < 16; ++e)
    f[e] = p[(size_t)(row0 + lr) * ld + koff + frag_k(e, half)];
  return f;
}

// A fragment from an LDS-resident f32 strip (16 rows x ldp floats)
__device__ __forceinline__ v16bf load_a_lds_f32(const float* p, int ldp,
                                                int koff, int lr, int half) {
  v16bf f;
#pragma unroll
  for (int e = 0; e < 16; ++e)
    f[e] = (__bf16)p[lr * ldp + koff + frag_k(e, half)];
  return f;
}

// B fragment (K=32, N=16) from row-major f32 B[k][n], leading dim ld
__device__ __forceinline__ v16bf load_b_f32(const float* __restrict__ p, int col0, int ld,
                                            int koff, int lr, int half) {
  v16bf f;
#pragma unroll
  for (int e = 0; e < 16; ++e)
    f[e] = (__bf16)p[(size_t)(koff + frag_k(e, half)) * ld + col0 + lr];
  return f;
}

// B fragment from row-major bf16
__device__ __forceinline__ v16bf load_b_bf16(const __bf16* __restrict__ p, int col0, int ld,
                                             int koff, int lr, int half) {
  v16bf f;
#pragma unroll
  for (int e = 0; e < 16; ++e)
    f[e] = p[(size_t)(koff + frag_k(e, half)) * ld + col0 + lr];
  return f;
}

// CDNA5 async global->LDS DMA of one 16-byte chunk per lane (ASYNCcnt-tracked)
__device__ __forceinline__ void async_g2l_b128(unsigned lds_byte_off, const void* gaddr) {
  asm volatile("global_load_async_to_lds_b128 %0, %1, off"
               :: "v"(lds_byte_off), "v"((unsigned long long)(uintptr_t)gaddr)
               : "memory");
}
__device__ __forceinline__ void wait_asynccnt0() {
  asm volatile("s_wait_asynccnt 0x0" ::: "memory");
}

// ============================================================================
// Kernel 1: Q/K/V projections. x[B*S, D] @ w[D, D] -> head-split bf16 [B,H,S,HD]
// grid = (8, 256, 3), block = 256 (8 waves; one 16x16 output tile per wave).
// The 16x1024 f32 x-strip is shared by all 8 waves: stage once via async DMA.
// ============================================================================
__global__ void qkv_proj_kernel(const float* __restrict__ x,
                                const float* __restrict__ wq,
                                const float* __restrict__ wk,
                                const float* __restrict__ wv,
                                __bf16* __restrict__ qws,
                                __bf16* __restrict__ kws,
                                __bf16* __restrict__ vws) {
  extern __shared__ float xs[];  // [16][kXP]
  const int tid = threadIdx.x;
  const int lane = tid & 31, wave = tid >> 5;
  const int lr = lane & 15, half = lane >> 4;
  const int col0 = (blockIdx.x * 8 + wave) * 16;  // N in [0,1024)
  const int row0 = blockIdx.y * 16;               // M in [0,4096)
  const int which = blockIdx.z;
  const float* w = (which == 0) ? wq : (which == 1) ? wk : wv;
  __bf16* dst    = (which == 0) ? qws : (which == 1) ? kws : vws;

  // ---- async-stage x strip: 4096 x b128 chunks across 256 threads ----
#pragma unroll
  for (int i = 0; i < 16; ++i) {
    int c = tid + 256 * i;          // chunk id
    int r = c >> 8;                 // row 0..15 (256 chunks per 1024-float row)
    int col = (c & 255) << 2;       // float col 0..1020
    async_g2l_b128((unsigned)((r * kXP + col) * sizeof(float)),
                   x + (size_t)(row0 + r) * kD + col);
  }
  wait_asynccnt0();
  __syncthreads();

  v8f c = {};
  for (int kb = 0; kb < kD; kb += 32) {
    v16bf a = load_a_lds_f32(xs, kXP, kb, lr, half);
    v16bf b = load_b_f32(w, col0, kD, kb, lr, half);
    c = wmma_bf16(a, b, c);
  }
#pragma unroll
  for (int vv = 0; vv < 8; ++vv) {
    int m = row0 + vv + 8 * half;       // token row in [0, B*S)
    int n = col0 + lr;                  // feature col in [0, D)
    int bb = m >> 11, ss = m & (kS - 1);
    int h = n >> 6, hd = n & (kHD - 1);
    dst[(((size_t)bb * kH + h) * kS + ss) * kHD + hd] = (__bf16)c[vv];
  }
}

// ============================================================================
// Kernel 2: per (b,h,16-row strip): scores -> masked softmax -> attn_weights,
// then P @ V -> attn_output.  grid = B*H*(S/16) = 4096, block = 128 (4 waves).
// Dynamic LDS: 16 rows x kPS floats (~128 KB of the 320 KB WGP LDS).
// ============================================================================
__global__ void attention_kernel(const __bf16* __restrict__ qws,
                                 const __bf16* __restrict__ kws,
                                 const __bf16* __restrict__ vws,
                                 const int* __restrict__ mask,
                                 float* __restrict__ out_w,
                                 float* __restrict__ out_a) {
  extern __shared__ float probs[];  // [16][kPS]
  const int lane = threadIdx.x & 31, wave = threadIdx.x >> 5;
  const int lr = lane & 15, half = lane >> 4;
  const int rt = blockIdx.x & 127;   // row tile within this head
  const int bh = blockIdx.x >> 7;    // b*H + h
  const int b  = bh >> 4;
  const int row0 = rt * 16;

  const __bf16* q = qws + (size_t)bh * kS * kHD;
  const __bf16* k = kws + (size_t)bh * kS * kHD;
  const __bf16* v = vws + (size_t)bh * kS * kHD;
  const float scale = 0.125f;  // 1/sqrt(HD=64)

  // ---- Phase 1: scores: Q(16xHD) @ K^T(HDx2048). Q fragments hoisted. ----
  const v16bf qf0 = load_a_bf16(q, row0, kHD, 0,  lr, half);
  const v16bf qf1 = load_a_bf16(q, row0, kHD, 32, lr, half);
  for (int ct = wave; ct < kS / 16; ct += 4) {
    v8f c = {};
    v16bf kt0 = load_a_bf16(k, ct * 16, kHD, 0,  lr, half);  // B[k,j]=K[j,k]: A-style load
    v16bf kt1 = load_a_bf16(k, ct * 16, kHD, 32, lr, half);
    c = wmma_bf16(qf0, kt0, c);
    c = wmma_bf16(qf1, kt1, c);
    int j  = ct * 16 + lr;
    int mk = mask[b * kS + j];
#pragma unroll
    for (int vv = 0; vv < 8; ++vv) {
      int r = vv + 8 * half;
      probs[r * kPS + j] = mk ? (c[vv] * scale) : -__builtin_inff();
    }
  }
  __syncthreads();

  // ---- Phase 2: row softmax (wave32 shuffle reductions), write attn_weights ----
  float* gw = out_w + ((size_t)bh * kS + row0) * kS;
  for (int r = wave * 4; r < wave * 4 + 4; ++r) {
    float mx = -__builtin_inff();
    for (int j = lane; j < kS; j += 32) mx = fmaxf(mx, probs[r * kPS + j]);
#pragma unroll
    for (int off = 16; off > 0; off >>= 1) mx = fmaxf(mx, __shfl_xor(mx, off, 32));

    float sum = 0.0f;
    for (int j = lane; j < kS; j += 32) {
      float e = __expf(probs[r * kPS + j] - mx);
      probs[r * kPS + j] = e;
      sum += e;
    }
#pragma unroll
    for (int off = 16; off > 0; off >>= 1) sum += __shfl_xor(sum, off, 32);

    float inv = 1.0f / sum;
    for (int j = lane; j < kS; j += 32) {
      float p = probs[r * kPS + j] * inv;
      probs[r * kPS + j] = p;
      gw[(size_t)r * kS + j] = p;
    }
  }
  __syncthreads();

  // ---- Phase 3: attn_output tile: P(16x2048) @ V(2048x64); one 16-col tile per wave ----
  {
    const int col0 = wave * 16;
    v8f c = {};
    for (int kb = 0; kb < kS; kb += 32) {
      v16bf a = load_a_lds_f32(probs, kPS, kb, lr, half);
      v16bf bb = load_b_bf16(v, col0, kHD, kb, lr, half);
      c = wmma_bf16(a, bb, c);
    }
#pragma unroll
    for (int vv = 0; vv < 8; ++vv) {
      int r = row0 + vv + 8 * half;
      out_a[((size_t)bh * kS + r) * kHD + col0 + lr] = c[vv];
    }
  }
}

// ============================================================================
// Kernel 3: output = merged(attn_output) @ wo. The head-merge gather is folded
// into the async-DMA staging addresses; inner loop: A from LDS, B (wo) from global, WMMA.
// grid = (8, 256), block = 256.
// ============================================================================
__global__ void out_proj_kernel(const float* __restrict__ attn,
                                const float* __restrict__ wo,
                                float* __restrict__ out) {
  extern __shared__ float xs[];  // [16][kXP] merged A strip
  const int tid = threadIdx.x;
  const int lane = tid & 31, wave = tid >> 5;
  const int lr = lane & 15, half = lane >> 4;
  const int col0 = (blockIdx.x * 8 + wave) * 16;
  const int row0 = blockIdx.y * 16;

  // ---- async-stage merged strip: chunk never crosses a 64-float head span ----
#pragma unroll
  for (int i = 0; i < 16; ++i) {
    int c = tid + 256 * i;
    int r = c >> 8;
    int col = (c & 255) << 2;       // merged feature col = h*HD + hd
    int h = col >> 6, hd = col & (kHD - 1);
    int m = row0 + r;
    int bb = m >> 11, ss = m & (kS - 1);
    async_g2l_b128((unsigned)((r * kXP + col) * sizeof(float)),
                   attn + (((size_t)bb * kH + h) * kS + ss) * kHD + hd);
  }
  wait_asynccnt0();
  __syncthreads();

  v8f c = {};
  for (int kb = 0; kb < kD; kb += 32) {
    v16bf a = load_a_lds_f32(xs, kXP, kb, lr, half);
    v16bf b = load_b_f32(wo, col0, kD, kb, lr, half);
    c = wmma_bf16(a, b, c);
  }
#pragma unroll
  for (int vv = 0; vv < 8; ++vv) {
    int mr = row0 + vv + 8 * half;
    out[(size_t)mr * kD + col0 + lr] = c[vv];
  }
}

// ============================================================================
extern "C" void kernel_launch(void* const* d_in, const int* in_sizes, int n_in,
                              void* d_out, int out_size, void* d_ws, size_t ws_size,
                              hipStream_t stream) {
  const float* x    = (const float*)d_in[0];
  const int*   mask = (const int*)d_in[1];
  const float* wq   = (const float*)d_in[2];
  const float* wk   = (const float*)d_in[3];
  const float* wv   = (const float*)d_in[4];
  const float* wo   = (const float*)d_in[5];

  float* out = (float*)d_out;
  const size_t n_out1 = (size_t)kB * kS * kD;          // output
  const size_t n_w    = (size_t)kB * kH * kS * kS;     // attn_weights
  float* out_w = out + n_out1;
  float* out_a = out_w + n_w;                          // attn_output [B,H,S,HD]

  // workspace: bf16 Q,K,V in [B,H,S,HD] (3 x 8 MB = 24 MB)
  const size_t per = (size_t)kB * kH * kS * kHD;
  __bf16* qws = (__bf16*)d_ws;
  __bf16* kws = qws + per;
  __bf16* vws = kws + per;

  const size_t strip_bytes = (size_t)16 * kXP * sizeof(float);  // ~64 KB
  const size_t attn_bytes  = (size_t)16 * kPS * sizeof(float);  // ~128 KB < 320 KB WGP LDS

  qkv_proj_kernel<<<dim3(8, 256, 3), 256, strip_bytes, stream>>>(x, wq, wk, wv, qws, kws, vws);
  attention_kernel<<<dim3(kB * kH * (kS / 16)), 128, attn_bytes, stream>>>(
      qws, kws, vws, mask, out_w, out_a);
  out_proj_kernel<<<dim3(8, 256), 256, strip_bytes, stream>>>(out_a, wo, out);
}
